// GraphConvolution_67353677136105
// MI455X (gfx1250) — matrix operand order repaired and checked by reading, not compiled, not run on hardware
//
#include <hip/hip_runtime.h>
#include <hip/hip_bf16.h>

typedef __attribute__((ext_vector_type(16))) _Float16 v16h;
typedef __attribute__((ext_vector_type(8)))  _Float16 v8h;
typedef __attribute__((ext_vector_type(8)))  float    v8f;

#define NN   4096
#define EE   131072
#define DD   256

// ---------------------------------------------------------------------------
// CDNA5 async memory->LDS helpers (ASYNCcnt-tracked, no VGPR round trip).
// LDS aperture: flat addr[31:0] == LDS byte offset (ISA 10.2), so truncating
// the generic pointer yields the VDST byte address.
// ---------------------------------------------------------------------------
__device__ __forceinline__ void async_b128(void* lds_dst, const void* gsrc) {
    asm volatile("global_load_async_to_lds_b128 %0, %1, off"
                 :: "v"((unsigned)(unsigned long long)lds_dst), "v"(gsrc)
                 : "memory");
}
__device__ __forceinline__ void async_b64(void* lds_dst, const void* gsrc) {
    asm volatile("global_load_async_to_lds_b64 %0, %1, off"
                 :: "v"((unsigned)(unsigned long long)lds_dst), "v"(gsrc)
                 : "memory");
}
__device__ __forceinline__ void wait_async0() {
    asm volatile("s_wait_asynccnt 0" ::: "memory");
}
__device__ __forceinline__ void wait_ds0() {
    asm volatile("s_wait_dscnt 0" ::: "memory");
}

// ---------------------------------------------------------------------------
// elementwise helpers
// ---------------------------------------------------------------------------
__global__ void k_cvt_f16(_Float16* __restrict__ dst, const float* __restrict__ src) {
    size_t i = (size_t)blockIdx.x * 256 + threadIdx.x;
    dst[i] = (_Float16)src[i];
}

__global__ void k_copy_f32(float* __restrict__ dst, const float* __restrict__ src) {
    size_t i = (size_t)blockIdx.x * 256 + threadIdx.x;
    dst[i] = src[i];
}

// dst[k*Nn + n] = src[n*K + k]   (src is [Nn][K] row-major)
__global__ void k_transcvt(_Float16* __restrict__ dst, const float* __restrict__ src,
                           int K, int Nn) {
    size_t i = (size_t)blockIdx.x * 256 + threadIdx.x;
    int k = (int)(i / Nn), n = (int)(i % Nn);
    dst[i] = (_Float16)src[(size_t)n * K + k];
}

// B_final [768][256]: rows 0..255 = W_l[0:256], 256..511 = W_g[0:256],
// 512..767 = W_l[256:512] + W_g[256:512]
__global__ void k_bfinal(const float* __restrict__ wl, const float* __restrict__ wg,
                         _Float16* __restrict__ bf) {
    size_t i = (size_t)blockIdx.x * 256 + threadIdx.x;   // 768*256
    int k = (int)(i >> 8), n = (int)(i & 255);
    float v;
    if (k < 256)      v = wl[(size_t)k * 256 + n];
    else if (k < 512) v = wg[(size_t)(k - 256) * 256 + n];
    else              v = wl[(size_t)(k - 256) * 256 + n] + wg[(size_t)(k - 256) * 256 + n];
    bf[i] = (_Float16)v;
}

// edge_attr[e][d] = dot(distance_map[e][:16], lin_w[d][:16]) + lin_b[d]  -> f16
__global__ void k_edge_attr(const float* __restrict__ dm, const float* __restrict__ w,
                            const float* __restrict__ b, _Float16* __restrict__ ea) {
    size_t i = (size_t)blockIdx.x * 256 + threadIdx.x;   // E*256
    int e = (int)(i >> 8), d = (int)(i & 255);
    const float* dr = dm + (size_t)e * 16;
    const float* wr = w  + (size_t)d * 16;
    float acc = b[d];
#pragma unroll
    for (int j = 0; j < 16; ++j) acc = fmaf(dr[j], wr[j], acc);
    ea[i] = (_Float16)acc;
}

// x_out[dst[e]][d] += norm[e] * (x[src[e]][d] + edge_attr[e][d])
__global__ void k_scatter(const float* __restrict__ x, const _Float16* __restrict__ ea,
                          const float* __restrict__ norm, const int* __restrict__ adj,
                          float* __restrict__ xout) {
    size_t i = (size_t)blockIdx.x * 256 + threadIdx.x;   // E*64 threads, 4 feat each
    int e  = (int)(i >> 6);
    int d4 = (int)(i & 63) * 4;
    int s  = adj[e], d = adj[EE + e];
    float nv = norm[e];
#pragma unroll
    for (int j = 0; j < 4; ++j) {
        float v = nv * (x[(size_t)s * DD + d4 + j] + (float)ea[(size_t)e * DD + d4 + j]);
        atomicAdd(&xout[(size_t)d * DD + d4 + j], v);
    }
}

// output = theta*G + (1-theta)*((1-a)*(h_local + h_global) + 2a*h0)
__global__ void k_combine(const float* __restrict__ G, const float* __restrict__ xout,
                          const _Float16* __restrict__ hg, const float* __restrict__ h0,
                          const float* __restrict__ alpha, const int* __restrict__ lamda,
                          const int* __restrict__ lp, float* __restrict__ out) {
    size_t i = (size_t)blockIdx.x * 256 + threadIdx.x;   // N*D
    float a = alpha[0];
    float theta = fminf(1.f, logf((float)lamda[0] / (float)lp[0] + 1.f));
    float r = (1.f - a) * (xout[i] + (float)hg[i]) + 2.f * a * h0[i];
    out[i] = theta * G[i] + (1.f - theta) * r;
}

// ---------------------------------------------------------------------------
// WMMA tiled GEMM: C[M,Nn] = gather(A)[M,K] * B[K,Nn] (+bias, epilogue)
// 128 threads = 4 waves; block tile 64 x NB; K step 32; A/B staged with
// GLOBAL_LOAD_ASYNC_TO_LDS (ASYNCcnt).
// ---------------------------------------------------------------------------
#define GM_DENSE 0
#define GM_EDGE  1
#define GM_CAT3  2
#define EPI_GELU_F16 0
#define EPI_BIAS_F32 1
#define EPI_BIAS_F16 2
#define EPI_RAW_F32  3

template <int GM, int EPI, int NB>
__global__ __launch_bounds__(128)
void gemm_wmma(const _Float16* __restrict__ A0, const _Float16* __restrict__ A1,
               const _Float16* __restrict__ A2,
               const int* __restrict__ srcI, const int* __restrict__ dstI,
               const _Float16* __restrict__ B, int K, int Nn,
               const float* __restrict__ bias,
               float* __restrict__ Cf, _Float16* __restrict__ Ch) {
    __shared__ _Float16 As[64 * 32];
    __shared__ _Float16 Bs[32 * NB];
    const int t    = threadIdx.x;
    const int wave = t >> 5;
    const int lane = t & 31;
    const int n16  = lane & 15;
    const int hi   = lane >> 4;
    const int m0   = blockIdx.x * 64;
    const int n0   = blockIdx.y * NB;

    v8f acc[NB / 16] = {};

    for (int k0 = 0; k0 < K; k0 += 32) {
        // ---- stage A tile (64x32) async: thread -> row t>>1, 16-half run ----
        {
            int row = t >> 1;
            int kk  = (t & 1) * 16;
            int e   = m0 + row;
            int kg  = k0 + kk;
            const _Float16* sp;
            if (GM == GM_DENSE) {
                sp = A0 + (size_t)e * K + kg;
            } else if (GM == GM_EDGE) {
                if (kg < 256)      sp = A0 + (size_t)srcI[e] * 256 + kg;
                else if (kg < 512) sp = A1 + (size_t)e * 256 + (kg - 256);
                else               sp = A2 + (size_t)dstI[e] * 256 + (kg - 512);
            } else {
                int seg = kg >> 8;
                const _Float16* base = (seg == 0) ? A0 : ((seg == 1) ? A1 : A2);
                sp = base + (size_t)e * 256 + (kg & 255);
            }
            async_b128(&As[row * 32 + kk], sp);
            async_b128(&As[row * 32 + kk + 8], sp + 8);
        }
        // ---- stage B tile (32xNB) async ----
        if (NB == 64) {
            int kk = t >> 2;
            int nn = (t & 3) * 16;
            const _Float16* s = B + (size_t)(k0 + kk) * Nn + n0 + nn;
            async_b128(&Bs[kk * 64 + nn], s);
            async_b128(&Bs[kk * 64 + nn + 8], s + 8);
        } else {  // NB == 16: 4 halves / thread
            int kk = t >> 2;
            int nn = (t & 3) * 4;
            const _Float16* s = B + (size_t)(k0 + kk) * Nn + n0 + nn;
            async_b64(&Bs[kk * 16 + nn], s);
        }
        if (k0 + 32 < K) __builtin_prefetch(B + (size_t)(k0 + 32) * Nn + n0, 0, 0);
        wait_async0();
        __syncthreads();

        // ---- fragments + WMMA ----
        v16h af;   // A 16x32: lane row = n16, K split by lane half
        {
            const _Float16* ap = &As[(wave * 16 + n16) * 32];
#pragma unroll
            for (int i = 0; i < 16; ++i)
                af[i] = ap[hi * 16 + i];
        }
#pragma unroll
        for (int nt = 0; nt < NB / 16; ++nt) {
            v16h bf;   // B 32x16: lane col = n16, K split by lane half
#pragma unroll
            for (int i = 0; i < 16; ++i)
                bf[i] = Bs[(hi * 16 + i) * NB + nt * 16 + n16];
            acc[nt] = __builtin_amdgcn_wmma_f32_16x16x32_f16(
                false, af, false, bf, (short)0, acc[nt], false, false);
        }
        wait_ds0();           // DS reads done before next iter's async overwrite
        __syncthreads();
    }

    // ---- epilogue: C frag lane holds col n16, rows hi*8 + r ----
#pragma unroll
    for (int nt = 0; nt < NB / 16; ++nt) {
#pragma unroll
        for (int r = 0; r < 8; ++r) {
            int row = m0 + wave * 16 + hi * 8 + r;
            int col = n0 + nt * 16 + n16;
            float v = acc[nt][r];
            if (EPI != EPI_RAW_F32) v += bias[col];
            if (EPI == EPI_GELU_F16) {
                v = 0.5f * v * (1.f + erff(v * 0.70710678118f));
                Ch[(size_t)row * Nn + col] = (_Float16)v;
            } else if (EPI == EPI_BIAS_F16) {
                Ch[(size_t)row * Nn + col] = (_Float16)v;
            } else {
                Cf[(size_t)row * Nn + col] = v;
            }
        }
    }
}

// ---------------------------------------------------------------------------
// Flash attention: one wave per (head, 16-query tile). HD=128, 2 heads.
// qkv layout [N,768] f16: q at +h*128, k at +256+h*128, v at +512+h*128.
// V tile staged to LDS via async-to-LDS; P routed through LDS to convert
// C-fragment layout -> A-fragment layout.
// ---------------------------------------------------------------------------
__global__ __launch_bounds__(32)
void k_attn(const _Float16* __restrict__ qkv, _Float16* __restrict__ out) {
    __shared__ _Float16 Vs[32 * 128];
    __shared__ _Float16 Ps[16 * 32];
    const int lane = threadIdx.x;
    const int n16  = lane & 15;
    const int hi   = lane >> 4;
    const int q0   = blockIdx.x * 16;
    const int head = blockIdx.y;
    const int qoff = head * 128;
    const int koff = 256 + head * 128;
    const int voff = 512 + head * 128;
    const float scale = 0.088388347648f;   // 1/sqrt(128)

    // Q fragments for the 4 K-steps of the d=128 contraction, kept resident
    v16h qf[4];
#pragma unroll
    for (int s = 0; s < 4; ++s) {
        const _Float16* qp = qkv + (size_t)(q0 + n16) * 768 + qoff + s * 32;
#pragma unroll
        for (int i = 0; i < 16; ++i)
            qf[s][i] = qp[hi * 16 + i];
    }
    v8f o[8] = {};
    float mx[8], sm[8];
#pragma unroll
    for (int r = 0; r < 8; ++r) { mx[r] = -1e30f; sm[r] = 0.f; }

    for (int kb = 0; kb < NN; kb += 32) {
        // stage V tile 32x128 to LDS asynchronously (lane copies its row)
        {
            const _Float16* vp = qkv + (size_t)(kb + lane) * 768 + voff;
#pragma unroll
            for (int i = 0; i < 16; ++i)
                async_b128(&Vs[lane * 128 + i * 8], vp + i * 8);
        }
        // S = Q K^T over d=128 : two 16x16 C tiles (keys 0..15 / 16..31)
        v8f sc[2] = {};
#pragma unroll
        for (int s = 0; s < 4; ++s) {
#pragma unroll
            for (int tt = 0; tt < 2; ++tt) {
                v16h kf;  // B frag: lane col = key n16, contiguous 16 d-values
                const _Float16* kp = qkv + (size_t)(kb + tt * 16 + n16) * 768
                                     + koff + s * 32 + hi * 16;
#pragma unroll
                for (int i = 0; i < 16; ++i) kf[i] = kp[i];
                sc[tt] = __builtin_amdgcn_wmma_f32_16x16x32_f16(
                    false, qf[s], false, kf, (short)0, sc[tt], false, false);
            }
        }
        // online softmax (row = hi*8 + r; reduce across the 16 lanes of a half)
#pragma unroll
        for (int r = 0; r < 8; ++r) {
            float s0 = sc[0][r] * scale, s1 = sc[1][r] * scale;
            float vmax = fmaxf(s0, s1);
            for (int m = 1; m < 16; m <<= 1) vmax = fmaxf(vmax, __shfl_xor(vmax, m, 32));
            float nm = fmaxf(mx[r], vmax);
            float f  = __expf(mx[r] - nm);
            float p0 = __expf(s0 - nm), p1 = __expf(s1 - nm);
            float rs = p0 + p1;
            for (int m = 1; m < 16; m <<= 1) rs += __shfl_xor(rs, m, 32);
            sm[r] = sm[r] * f + rs;
            mx[r] = nm;
#pragma unroll
            for (int nt = 0; nt < 8; ++nt) o[nt][r] *= f;
            int prow = hi * 8 + r;
            Ps[prow * 32 + n16]      = (_Float16)p0;
            Ps[prow * 32 + 16 + n16] = (_Float16)p1;
        }
        wait_async0();   // V tile resident in LDS
        wait_ds0();      // P stores visible
        // P (A frag) @ V (B frags): accumulate O 16x128
        v16h pa;
        {
            const _Float16* pp = &Ps[n16 * 32];
#pragma unroll
            for (int i = 0; i < 16; ++i)
                pa[i] = pp[hi * 16 + i];
        }
#pragma unroll
        for (int nt = 0; nt < 8; ++nt) {
            v16h vb;
#pragma unroll
            for (int i = 0; i < 16; ++i)
                vb[i] = Vs[(hi * 16 + i) * 128 + nt * 16 + n16];
            o[nt] = __builtin_amdgcn_wmma_f32_16x16x32_f16(
                false, pa, false, vb, (short)0, o[nt], false, false);
        }
        wait_ds0();      // DS reads of Vs done before next iter's async overwrite
    }
#pragma unroll
    for (int nt = 0; nt < 8; ++nt)
#pragma unroll
        for (int r = 0; r < 8; ++r) {
            float val = o[nt][r] / sm[r];
            out[(size_t)(q0 + hi * 8 + r) * 256 + head * 128 + nt * 16 + n16] =
                (_Float16)val;
        }
}

// ---------------------------------------------------------------------------
extern "C" void kernel_launch(void* const* d_in, const int* in_sizes, int n_in,
                              void* d_out, int out_size, void* d_ws, size_t ws_size,
                              hipStream_t stream) {
    const float* input  = (const float*)d_in[0];
    const int*   adj    = (const int*)d_in[1];
    const float* h0     = (const float*)d_in[2];
    const int*   lamda  = (const int*)d_in[3];
    const float* alpha  = (const float*)d_in[4];
    const int*   lparam = (const int*)d_in[5];
    const float* dmap   = (const float*)d_in[6];
    const float* norm   = (const float*)d_in[7];
    const float* wl     = (const float*)d_in[8];
    const float* wg     = (const float*)d_in[9];
    const float* lin_w  = (const float*)d_in[10];
    const float* lin_b  = (const float*)d_in[11];
    const float* w1     = (const float*)d_in[12];
    const float* b1     = (const float*)d_in[13];
    const float* w2     = (const float*)d_in[14];
    const float* b2     = (const float*)d_in[15];
    const float* w3     = (const float*)d_in[16];
    const float* b3     = (const float*)d_in[17];
    const float* wi     = (const float*)d_in[18];
    const float* bi     = (const float*)d_in[19];
    const float* wo     = (const float*)d_in[20];
    const float* bo     = (const float*)d_in[21];

    float* out_main = (float*)d_out;                        // [N,256]
    float* out_dm   = out_main + (size_t)NN * DD;           // [E,16]

    // carve workspace
    char* ws = (char*)d_ws;
    auto carve = [&](size_t bytes) {
        char* p = ws;
        ws += (bytes + 255) & ~(size_t)255;
        return p;
    };
    _Float16* w1T   = (_Float16*)carve((size_t)768 * 512 * 2);
    _Float16* w2T   = (_Float16*)carve((size_t)512 * 256 * 2);
    _Float16* w3T   = (_Float16*)carve((size_t)256 * 16 * 2);
    _Float16* wiT   = (_Float16*)carve((size_t)256 * 768 * 2);
    _Float16* woT   = (_Float16*)carve((size_t)256 * 256 * 2);
    _Float16* bfin  = (_Float16*)carve((size_t)768 * 256 * 2);
    _Float16* in_h  = (_Float16*)carve((size_t)NN * DD * 2);
    _Float16* h0_h  = (_Float16*)carve((size_t)NN * DD * 2);
    float*    xout  = (float*)   carve((size_t)NN * DD * 4);
    _Float16* xo_h  = (_Float16*)carve((size_t)NN * DD * 2);
    _Float16* ea_h  = (_Float16*)carve((size_t)EE * DD * 2);   // reused as h2
    _Float16* h1_h  = (_Float16*)carve((size_t)EE * 512 * 2);
    _Float16* qkv_h = (_Float16*)carve((size_t)NN * 768 * 2);
    _Float16* at_h  = (_Float16*)carve((size_t)NN * DD * 2);
    _Float16* hg_h  = (_Float16*)carve((size_t)NN * DD * 2);
    float*    Gbuf  = (float*)   carve((size_t)NN * DD * 4);
    _Float16* h2_h  = ea_h;   // edge_attr dead after the h1 GEMM

    const int* srcI = adj;
    const int* dstI = adj + EE;

    // --- weight prep / casts ---
    k_cvt_f16<<<(NN * DD) / 256, 256, 0, stream>>>(in_h, input);
    k_cvt_f16<<<(NN * DD) / 256, 256, 0, stream>>>(h0_h, h0);
    k_transcvt<<<(768 * 512) / 256, 256, 0, stream>>>(w1T, w1, 768, 512);
    k_transcvt<<<(512 * 256) / 256, 256, 0, stream>>>(w2T, w2, 512, 256);
    k_transcvt<<<(256 * 16) / 256, 256, 0, stream>>>(w3T, w3, 256, 16);
    k_transcvt<<<(256 * 768) / 256, 256, 0, stream>>>(wiT, wi, 256, 768);
    k_transcvt<<<(256 * 256) / 256, 256, 0, stream>>>(woT, wo, 256, 256);
    k_bfinal<<<(768 * 256) / 256, 256, 0, stream>>>(wl, wg, bfin);

    // --- EA_GCN aggregation ---
    k_edge_attr<<<EE, 256, 0, stream>>>(dmap, lin_w, lin_b, ea_h);
    k_copy_f32<<<(NN * DD) / 256, 256, 0, stream>>>(xout, input);
    k_scatter<<<(EE * 64) / 256, 256, 0, stream>>>(input, ea_h, norm, adj, xout);
    k_cvt_f16<<<(NN * DD) / 256, 256, 0, stream>>>(xo_h, xout);

    // --- edge MLP (dominant GEMMs) ---
    gemm_wmma<GM_EDGE, EPI_GELU_F16, 64><<<dim3(EE / 64, 512 / 64), 128, 0, stream>>>(
        xo_h, ea_h, xo_h, srcI, dstI, w1T, 768, 512, b1, nullptr, h1_h);
    gemm_wmma<GM_DENSE, EPI_GELU_F16, 64><<<dim3(EE / 64, 256 / 64), 128, 0, stream>>>(
        h1_h, nullptr, nullptr, nullptr, nullptr, w2T, 512, 256, b2, nullptr, h2_h);
    gemm_wmma<GM_DENSE, EPI_BIAS_F32, 16><<<dim3(EE / 64, 1), 128, 0, stream>>>(
        h2_h, nullptr, nullptr, nullptr, nullptr, w3T, 256, 16, b3, out_dm, nullptr);

    // --- MHA ---
    gemm_wmma<GM_DENSE, EPI_BIAS_F16, 64><<<dim3(NN / 64, 768 / 64), 128, 0, stream>>>(
        in_h, nullptr, nullptr, nullptr, nullptr, wiT, 256, 768, bi, nullptr, qkv_h);
    k_attn<<<dim3(NN / 16, 2), 32, 0, stream>>>(qkv_h, at_h);
    gemm_wmma<GM_DENSE, EPI_BIAS_F16, 64><<<dim3(NN / 64, 256 / 64), 128, 0, stream>>>(
        at_h, nullptr, nullptr, nullptr, nullptr, woT, 256, 256, bo, nullptr, hg_h);

    // --- fused final GEMM: [h_local | h_global | h0] (K=768) @ B_final ---
    gemm_wmma<GM_CAT3, EPI_RAW_F32, 64><<<dim3(NN / 64, 256 / 64), 128, 0, stream>>>(
        xo_h, hg_h, h0_h, nullptr, nullptr, bfin, 768, 256, nullptr, Gbuf, nullptr);
    k_combine<<<(NN * DD) / 256, 256, 0, stream>>>(Gbuf, xout, hg_h, h0,
                                                   alpha, lamda, lparam, out_main);
}
